// MiniMaxText01DecoderLayer_64244120813655
// MI455X (gfx1250) — compile-verified, matrix operand order checked by reading, not run on hardware
//
#include <hip/hip_runtime.h>
#include <hip/hip_bf16.h>
#include <math.h>

// ---------------------------------------------------------------------------
// MiniMax-Text-01 decoder layer for gfx1250 (MI455X).
// All matmuls via v_wmma_f32_16x16x32_bf16 (wave32, f32 accumulate).
// LDS fragments loaded as ds_load_b128; global staging as b128 loads.
// ---------------------------------------------------------------------------

typedef __attribute__((ext_vector_type(16))) __bf16 v16bf;
typedef __attribute__((ext_vector_type(8)))  float  v8f;

#define T_TOK 1024
#define HID   2048
#define NH    16
#define HD    128
#define QKVN  6144
#define FFN   2048
#define NEXP  8
#define NBLK  4
#define EPS_LN   1e-5f
#define EPS_ATTN 1.1920929e-07f

enum { EPI_NONE = 0, EPI_SILU = 1, EPI_SIGMOID = 2, EPI_RESADD = 3, EPI_ROWSCALE = 4 };

__device__ __forceinline__ v8f v8zero() {
  v8f z;
#pragma unroll
  for (int i = 0; i < 8; ++i) z[i] = 0.0f;
  return z;
}

__device__ __forceinline__ v8f wmma_bf16(v16bf a, v16bf b, v8f c) {
  return __builtin_amdgcn_wmma_f32_16x16x32_bf16(false, a, false, b, (short)0, c, false, false);
}

union Frag16 { v16bf v; uint4 q[2]; };
union Pack4  { __bf16 h[4]; uint2 u; };

__device__ __forceinline__ void st_bf16x4(__bf16* dst, float a, float b, float c, float d) {
  Pack4 p;
  p.h[0] = (__bf16)a; p.h[1] = (__bf16)b; p.h[2] = (__bf16)c; p.h[3] = (__bf16)d;
  *(uint2*)dst = p.u;
}

// Fragment from row-major bf16 LDS tile, element [outer, k] = tile[row*ldr + k].
// K pattern: elems 0..7 -> k = half*8 + 0..7 ; elems 8..15 -> k = 16 + half*8 + 0..7.
// Both groups are 8 contiguous bf16 (16B) -> two ds_load_b128.
__device__ __forceinline__ v16bf ld_frag_row(const __bf16* tile, int row_base, int ldr, int k_base) {
  const int lane = threadIdx.x & 31;
  const int half = lane >> 4;
  const uint4* p = (const uint4*)(tile + (row_base + (lane & 15)) * ldr + k_base + half * 8);
  Frag16 f;
  f.q[0] = p[0];   // k = k_base + half*8      .. +7
  f.q[1] = p[2];   // k = k_base + 16 + half*8 .. +7
  return f.v;
}

// Transposed-index fragment: element [outer, k] = tile[k*ldr + outer].
__device__ __forceinline__ v16bf ld_frag_col(const __bf16* tile, int row_base, int ldr, int k_base) {
  const int lane = threadIdx.x & 31;
  const int half = lane >> 4;
  const __bf16* cp = tile + (row_base + (lane & 15));
  v16bf f;
#pragma unroll
  for (int j = 0; j < 8; ++j) {
    int k0 = k_base + ((j < 4) ? (half * 8 + 2 * j) : (16 + half * 8 + 2 * (j - 4)));
    f[2 * j]     = cp[(size_t)k0 * ldr];
    f[2 * j + 1] = cp[(size_t)(k0 + 1) * ldr];
  }
  return f;
}

// A fragment from a row-major fp32 global matrix (b128 loads + cvt to bf16).
__device__ __forceinline__ v16bf ld_frag_g32(const float* base, int row_base, int ldr, int k_base) {
  const int lane = threadIdx.x & 31;
  const int half = lane >> 4;
  const float* rp = base + (size_t)(row_base + (lane & 15)) * ldr + k_base + half * 8;
  float4 a0 = ((const float4*)rp)[0];
  float4 a1 = ((const float4*)rp)[1];
  float4 b0 = ((const float4*)(rp + 16))[0];
  float4 b1 = ((const float4*)(rp + 16))[1];
  v16bf f;
  f[0] = (__bf16)a0.x; f[1] = (__bf16)a0.y; f[2] = (__bf16)a0.z; f[3] = (__bf16)a0.w;
  f[4] = (__bf16)a1.x; f[5] = (__bf16)a1.y; f[6] = (__bf16)a1.z; f[7] = (__bf16)a1.w;
  f[8]  = (__bf16)b0.x; f[9]  = (__bf16)b0.y; f[10] = (__bf16)b0.z; f[11] = (__bf16)b0.w;
  f[12] = (__bf16)b1.x; f[13] = (__bf16)b1.y; f[14] = (__bf16)b1.z; f[15] = (__bf16)b1.w;
  return f;
}

// ---------------------------------------------------------------------------
// Generic WMMA GEMM:  C[M,N] = epi( A[M,K] @ B[N,K]^T )
// Block tile 128x128x32, 8 waves of 32x64; grouped/gather options for MoE.
// ---------------------------------------------------------------------------
template <int EPI, bool A_BF16, bool GROUPED, bool GATHER>
__global__ __launch_bounds__(256) void gemm_kernel(
    const void* __restrict__ Aptr, const float* __restrict__ Ball, float* __restrict__ C,
    int M, int N, int K, int lda, int ldc,
    const float* __restrict__ aux, const int* __restrict__ gidx,
    const int* __restrict__ counts, const int* __restrict__ offs, long long strideB) {
  const int e = GROUPED ? blockIdx.z : 0;
  const int rcount = GROUPED ? counts[e] : M;
  const int rbase  = GROUPED ? offs[e] : 0;
  if ((int)blockIdx.y * 128 >= rcount) return;
  const float* B = Ball + (size_t)e * (size_t)strideB;

  __shared__ __attribute__((aligned(16))) __bf16 As[128 * 32];
  __shared__ __attribute__((aligned(16))) __bf16 Bs[128 * 32];

  const int tid = threadIdx.x, lane = tid & 31, wid = tid >> 5;
  const int wm = wid & 3, wn = wid >> 2;
  const int nblock = blockIdx.x * 128;
  const int srow = tid >> 3;        // 0..31
  const int scol = (tid & 7) * 4;   // 0..28

  v8f acc[2][4];
#pragma unroll
  for (int i = 0; i < 2; ++i)
#pragma unroll
    for (int j = 0; j < 4; ++j) acc[i][j] = v8zero();

  for (int k0 = 0; k0 < K; k0 += 32) {
    __syncthreads();
#pragma unroll
    for (int i = 0; i < 4; ++i) {
      int row = i * 32 + srow;
      // ---- stage A (fp32 or bf16 -> bf16), b128/b64 loads, ds_store_b64
      int gr = blockIdx.y * 128 + row;
      float4 av = make_float4(0.f, 0.f, 0.f, 0.f);
      if (gr < rcount) {
        int arow = rbase + gr;
        int tok = GATHER ? gidx[arow] : arow;
        if (A_BF16) {
          Pack4 t;
          t.u = *(const uint2*)((const __bf16*)Aptr + (size_t)tok * lda + k0 + scol);
          av = make_float4((float)t.h[0], (float)t.h[1], (float)t.h[2], (float)t.h[3]);
        } else {
          av = *(const float4*)((const float*)Aptr + (size_t)tok * lda + k0 + scol);
        }
      }
      st_bf16x4(&As[row * 32 + scol], av.x, av.y, av.z, av.w);
      // ---- stage B (weights fp32 -> bf16)
      int nrow = nblock + row;
      float4 bv = make_float4(0.f, 0.f, 0.f, 0.f);
      if (nrow < N) bv = *(const float4*)(&B[(size_t)nrow * K + k0 + scol]);
      st_bf16x4(&Bs[row * 32 + scol], bv.x, bv.y, bv.z, bv.w);
      // ---- prefetch next B k-panel (global_prefetch_b8)
      if (k0 + 32 < K && nrow < N)
        __builtin_prefetch(&B[(size_t)nrow * K + k0 + 32 + scol], 0, 1);
    }
    __syncthreads();
    v16bf a0 = ld_frag_row(As, wm * 32, 32, 0);
    v16bf a1 = ld_frag_row(As, wm * 32 + 16, 32, 0);
#pragma unroll
    for (int fn = 0; fn < 4; ++fn) {
      v16bf bb = ld_frag_row(Bs, wn * 64 + fn * 16, 32, 0);
      acc[0][fn] = wmma_bf16(a0, bb, acc[0][fn]);
      acc[1][fn] = wmma_bf16(a1, bb, acc[1][fn]);
    }
  }

#pragma unroll
  for (int fm = 0; fm < 2; ++fm)
#pragma unroll
    for (int fn = 0; fn < 4; ++fn)
#pragma unroll
      for (int r = 0; r < 8; ++r) {
        int ml = wm * 32 + fm * 16 + (lane >> 4) * 8 + r;
        int gr = blockIdx.y * 128 + ml;
        if (gr >= rcount) continue;
        int crow = rbase + gr;
        int n = nblock + wn * 64 + fn * 16 + (lane & 15);
        float v = acc[fm][fn][r];
        if (EPI == EPI_SILU)     v = v / (1.0f + __expf(-v));
        if (EPI == EPI_SIGMOID)  v = 1.0f / (1.0f + __expf(-v));
        if (EPI == EPI_RESADD)   v += aux[(size_t)crow * ldc + n];
        if (EPI == EPI_ROWSCALE) v *= aux[crow];
        C[(size_t)crow * ldc + n] = v;
      }
}

// ---------------------------------------------------------------------------
// Lightning attention.  qkv layout: [token][h*384 + {q:0,k:128,v:256} + d]
// ---------------------------------------------------------------------------

// KV_b[d,e] = sum_m kdec[m]*k[m,d]*v[m,e]  per (head, 256-token block)
__global__ __launch_bounds__(256) void attn_kv_kernel(const float* __restrict__ qkv,
                                                      const float* __restrict__ slopes,
                                                      float* __restrict__ kvb) {
  const int b = blockIdx.x, h = blockIdx.y;
  const float s = slopes[h];
  __shared__ __attribute__((aligned(16))) __bf16 kt[32 * 128];
  __shared__ __attribute__((aligned(16))) __bf16 vt[32 * 128];
  const int tid = threadIdx.x, lane = tid & 31, wid = tid >> 5;
  const int wm = wid & 3, wn = wid >> 2;

  v8f acc[2][4];
#pragma unroll
  for (int i = 0; i < 2; ++i)
#pragma unroll
    for (int j = 0; j < 4; ++j) acc[i][j] = v8zero();

  for (int mb = 0; mb < 8; ++mb) {
    __syncthreads();
#pragma unroll
    for (int i = 0; i < 4; ++i) {
      int row = i * 8 + (tid >> 5);
      int col4 = (tid & 31) * 4;
      int m = mb * 32 + row;
      const float* rp = qkv + (size_t)(b * 256 + m) * QKVN + h * 384;
      float kd = __expf(-s * (float)(255 - m));   // exp(-s*(BLOCK-(m+1)))
      float4 k4 = *(const float4*)(rp + 128 + col4);
      float4 v4 = *(const float4*)(rp + 256 + col4);
      st_bf16x4(&kt[row * 128 + col4], k4.x * kd, k4.y * kd, k4.z * kd, k4.w * kd);
      st_bf16x4(&vt[row * 128 + col4], v4.x, v4.y, v4.z, v4.w);
    }
    __syncthreads();
    v16bf a0 = ld_frag_col(kt, wm * 32, 128, 0);        // A[d, m] = kdec[m,d]
    v16bf a1 = ld_frag_col(kt, wm * 32 + 16, 128, 0);
#pragma unroll
    for (int fn = 0; fn < 4; ++fn) {
      v16bf bb = ld_frag_col(vt, wn * 64 + fn * 16, 128, 0);  // B[e, m] = v[m,e]
      acc[0][fn] = wmma_bf16(a0, bb, acc[0][fn]);
      acc[1][fn] = wmma_bf16(a1, bb, acc[1][fn]);
    }
  }
  float* out = kvb + (size_t)(h * NBLK + b) * (HD * HD);
#pragma unroll
  for (int fm = 0; fm < 2; ++fm)
#pragma unroll
    for (int fn = 0; fn < 4; ++fn)
#pragma unroll
      for (int r = 0; r < 8; ++r) {
        int d  = wm * 32 + fm * 16 + (lane >> 4) * 8 + r;
        int ee = wn * 64 + fn * 16 + (lane & 15);
        out[(size_t)d * HD + ee] = acc[fm][fn][r];
      }
}

// kv_state_b = sum_{j<b} block_decay^{b-1-j} * KV_j
__global__ __launch_bounds__(256) void attn_prefix_kernel(const float* __restrict__ kvb,
                                                          const float* __restrict__ slopes,
                                                          float* __restrict__ kvs) {
  const int b = blockIdx.x, h = blockIdx.y;
  const float bd = __expf(-slopes[h] * 256.0f);
  float* dst = kvs + (size_t)(h * NBLK + b) * (HD * HD);
  for (int i = threadIdx.x; i < HD * HD; i += 256) {
    float v = 0.0f;
    for (int j = 0; j < b; ++j)
      v = v * bd + kvb[(size_t)(h * NBLK + j) * (HD * HD) + i];
    dst[i] = v;
  }
}

// out = rowscale_qdec(q @ kv_state) + (tril_decay(q @ k^T)) @ v
__global__ __launch_bounds__(256) void attn_out_kernel(const float* __restrict__ qkv,
                                                       const float* __restrict__ kvs,
                                                       const float* __restrict__ slopes,
                                                       float* __restrict__ attn) {
  const int b = blockIdx.x, h = blockIdx.y;
  const float s = slopes[h];
  __shared__ __attribute__((aligned(16))) __bf16 kvbf[HD * HD];     // 32 KB
  __shared__ __attribute__((aligned(16))) __bf16 kch[32 * 128];     //  8 KB
  __shared__ __attribute__((aligned(16))) __bf16 vch[32 * 128];     //  8 KB
  __shared__ __attribute__((aligned(16))) __bf16 sbuf[8 * 32 * 32]; // 16 KB
  const int tid = threadIdx.x, lane = tid & 31, wid = tid >> 5;

  const float* kvsrc = kvs + (size_t)(h * NBLK + b) * (HD * HD);
#pragma unroll
  for (int i = 0; i < 16; ++i) {
    int idx = (i * 256 + tid) * 4;
    float4 v4 = *(const float4*)(kvsrc + idx);
    st_bf16x4(&kvbf[idx], v4.x, v4.y, v4.z, v4.w);
  }
  __syncthreads();

  // cache raw q fragments for this wave's 32 rows (K-dim = d, 4 k-frags)
  const float* qbase = qkv + (size_t)(b * 256) * QKVN + h * 384;
  v16bf qf[2][4];
#pragma unroll
  for (int fm = 0; fm < 2; ++fm)
#pragma unroll
    for (int kf = 0; kf < 4; ++kf)
      qf[fm][kf] = ld_frag_g32(qbase, wid * 32 + fm * 16, QKVN, kf * 32);

  // o_nd = q @ kv_state (row-scale by q_decay afterwards: commutes)
  v8f acc[2][8];
#pragma unroll
  for (int i = 0; i < 2; ++i)
#pragma unroll
    for (int j = 0; j < 8; ++j) acc[i][j] = v8zero();
#pragma unroll
  for (int kf = 0; kf < 4; ++kf)
#pragma unroll
    for (int fn = 0; fn < 8; ++fn) {
      v16bf bb = ld_frag_col(kvbf, fn * 16, HD, kf * 32);   // B[e, d] = kv[d,e]
      acc[0][fn] = wmma_bf16(qf[0][kf], bb, acc[0][fn]);
      acc[1][fn] = wmma_bf16(qf[1][kf], bb, acc[1][fn]);
    }
#pragma unroll
  for (int fm = 0; fm < 2; ++fm)
#pragma unroll
    for (int fn = 0; fn < 8; ++fn)
#pragma unroll
      for (int r = 0; r < 8; ++r) {
        int m = wid * 32 + fm * 16 + (lane >> 4) * 8 + r;
        acc[fm][fn][r] *= __expf(-s * (float)(m + 1));      // q_decay
      }

  // intra-block: S = tril_decay(q @ k^T); out += S @ v, in chunks of 32 tokens
  __bf16* swave = sbuf + wid * (32 * 32);
  for (int nb = 0; nb < 8; ++nb) {
    __syncthreads();
#pragma unroll
    for (int i = 0; i < 4; ++i) {
      int row = i * 8 + (tid >> 5);
      int col4 = (tid & 31) * 4;
      const float* rp = qkv + (size_t)(b * 256 + nb * 32 + row) * QKVN + h * 384;
      float4 k4 = *(const float4*)(rp + 128 + col4);
      float4 v4 = *(const float4*)(rp + 256 + col4);
      st_bf16x4(&kch[row * 128 + col4], k4.x, k4.y, k4.z, k4.w);
      st_bf16x4(&vch[row * 128 + col4], v4.x, v4.y, v4.z, v4.w);
    }
    __syncthreads();

    v8f sfr[2][2];
#pragma unroll
    for (int i = 0; i < 2; ++i)
#pragma unroll
      for (int j = 0; j < 2; ++j) sfr[i][j] = v8zero();
#pragma unroll
    for (int kf = 0; kf < 4; ++kf)
#pragma unroll
      for (int fn = 0; fn < 2; ++fn) {
        v16bf bb = ld_frag_row(kch, fn * 16, 128, kf * 32); // B[n_tok, d] = k[n,d]
        sfr[0][fn] = wmma_bf16(qf[0][kf], bb, sfr[0][fn]);
        sfr[1][fn] = wmma_bf16(qf[1][kf], bb, sfr[1][fn]);
      }
    // causal diag decay + stage as A-operand for S@V (per-wave LDS, in-order DS)
#pragma unroll
    for (int fm = 0; fm < 2; ++fm)
#pragma unroll
      for (int fn = 0; fn < 2; ++fn)
#pragma unroll
        for (int r = 0; r < 8; ++r) {
          int ml = fm * 16 + (lane >> 4) * 8 + r;
          int m = wid * 32 + ml;
          int nl = fn * 16 + (lane & 15);
          int n = nb * 32 + nl;
          float v = sfr[fm][fn][r];
          v = (m >= n) ? v * __expf(-s * (float)(m - n)) : 0.0f;
          swave[ml * 32 + nl] = (__bf16)v;
        }
    v16bf sa0 = ld_frag_row(swave, 0, 32, 0);
    v16bf sa1 = ld_frag_row(swave, 16, 32, 0);
#pragma unroll
    for (int fn = 0; fn < 8; ++fn) {
      v16bf bb = ld_frag_col(vch, fn * 16, 128, 0);         // B[d_out, tok] = v[tok,d]
      acc[0][fn] = wmma_bf16(sa0, bb, acc[0][fn]);
      acc[1][fn] = wmma_bf16(sa1, bb, acc[1][fn]);
    }
  }

  // store: attn[token][h*128 + d]   (b h n d -> b n (h d))
#pragma unroll
  for (int fm = 0; fm < 2; ++fm)
#pragma unroll
    for (int fn = 0; fn < 8; ++fn)
#pragma unroll
      for (int r = 0; r < 8; ++r) {
        int m = wid * 32 + fm * 16 + (lane >> 4) * 8 + r;
        int n = fn * 16 + (lane & 15);
        attn[(size_t)(b * 256 + m) * HID + h * HD + n] = acc[fm][fn][r];
      }
}

// ---------------------------------------------------------------------------
// Elementwise / normalization / routing kernels (float4 vectorized)
// ---------------------------------------------------------------------------

__global__ __launch_bounds__(256) void rmsnorm_kernel(const float* __restrict__ x,
                                                      const float* __restrict__ w,
                                                      float eps, float* __restrict__ y) {
  __shared__ float red[256];
  const int row = blockIdx.x, tid = threadIdx.x;
  const float* xr = x + (size_t)row * HID;
  float4 loc[2];
  float ss = 0.0f;
#pragma unroll
  for (int i = 0; i < 2; ++i) {
    loc[i] = *(const float4*)(xr + tid * 4 + i * 1024);
    ss += loc[i].x * loc[i].x + loc[i].y * loc[i].y + loc[i].z * loc[i].z + loc[i].w * loc[i].w;
  }
  red[tid] = ss; __syncthreads();
  for (int st = 128; st > 0; st >>= 1) { if (tid < st) red[tid] += red[tid + st]; __syncthreads(); }
  float inv = rsqrtf(red[0] / (float)HID + eps);
#pragma unroll
  for (int i = 0; i < 2; ++i) {
    int c = tid * 4 + i * 1024;
    float4 wv = *(const float4*)(w + c);
    *(float4*)(y + (size_t)row * HID + c) =
        make_float4(loc[i].x * inv * wv.x, loc[i].y * inv * wv.y,
                    loc[i].z * inv * wv.z, loc[i].w * inv * wv.w);
  }
}

__global__ __launch_bounds__(256) void attn_gate_kernel(const float* __restrict__ attn,
                                                        const float* __restrict__ aw,
                                                        const float* __restrict__ gate,
                                                        float* __restrict__ m_in) {
  __shared__ float red[256];
  const int row = blockIdx.x, tid = threadIdx.x;
  const float* xr = attn + (size_t)row * HID;
  float4 loc[2];
  float ss = 0.0f;
#pragma unroll
  for (int i = 0; i < 2; ++i) {
    loc[i] = *(const float4*)(xr + tid * 4 + i * 1024);
    ss += loc[i].x * loc[i].x + loc[i].y * loc[i].y + loc[i].z * loc[i].z + loc[i].w * loc[i].w;
  }
  red[tid] = ss; __syncthreads();
  for (int st = 128; st > 0; st >>= 1) { if (tid < st) red[tid] += red[tid + st]; __syncthreads(); }
  float inv = rsqrtf(red[0] / (float)HID + EPS_ATTN);
#pragma unroll
  for (int i = 0; i < 2; ++i) {
    int c = tid * 4 + i * 1024;
    float4 wv = *(const float4*)(aw + c);
    float4 gv = *(const float4*)(gate + (size_t)row * HID + c);
    *(float4*)(m_in + (size_t)row * HID + c) =
        make_float4(loc[i].x * inv * wv.x * gv.x, loc[i].y * inv * wv.y * gv.y,
                    loc[i].z * inv * wv.z * gv.z, loc[i].w * inv * wv.w * gv.w);
  }
}

__global__ __launch_bounds__(256) void ln2_router_kernel(const float* __restrict__ hres,
                                                         const float* __restrict__ w,
                                                         const float* __restrict__ rw,
                                                         float* __restrict__ x2,
                                                         int* __restrict__ sel,
                                                         float* __restrict__ selw,
                                                         int* __restrict__ counts) {
  __shared__ float red[256];
  __shared__ float logits[NEXP];
  const int row = blockIdx.x, tid = threadIdx.x;
  const float* xr = hres + (size_t)row * HID;
  float4 loc[2];
  float ss = 0.0f;
#pragma unroll
  for (int i = 0; i < 2; ++i) {
    loc[i] = *(const float4*)(xr + tid * 4 + i * 1024);
    ss += loc[i].x * loc[i].x + loc[i].y * loc[i].y + loc[i].z * loc[i].z + loc[i].w * loc[i].w;
  }
  red[tid] = ss; __syncthreads();
  for (int st = 128; st > 0; st >>= 1) { if (tid < st) red[tid] += red[tid + st]; __syncthreads(); }
  float inv = rsqrtf(red[0] / (float)HID + EPS_LN);
  __syncthreads();
  float4 xs[2];
#pragma unroll
  for (int i = 0; i < 2; ++i) {
    int c = tid * 4 + i * 1024;
    float4 wv = *(const float4*)(w + c);
    xs[i] = make_float4(loc[i].x * inv * wv.x, loc[i].y * inv * wv.y,
                        loc[i].z * inv * wv.z, loc[i].w * inv * wv.w);
    *(float4*)(x2 + (size_t)row * HID + c) = xs[i];
  }
  for (int e = 0; e < NEXP; ++e) {
    float p = 0.0f;
#pragma unroll
    for (int i = 0; i < 2; ++i) {
      float4 rv = *(const float4*)(rw + (size_t)e * HID + tid * 4 + i * 1024);
      p += xs[i].x * rv.x + xs[i].y * rv.y + xs[i].z * rv.z + xs[i].w * rv.w;
    }
    red[tid] = p; __syncthreads();
    for (int st = 128; st > 0; st >>= 1) { if (tid < st) red[tid] += red[tid + st]; __syncthreads(); }
    if (tid == 0) logits[e] = red[0];
    __syncthreads();
  }
  if (tid == 0) {
    float mx = logits[0];
    for (int e = 1; e < NEXP; ++e) mx = fmaxf(mx, logits[e]);
    float ex[NEXP];
    for (int e = 0; e < NEXP; ++e) ex[e] = __expf(logits[e] - mx);
    int i0 = 0;
    for (int e = 1; e < NEXP; ++e) if (ex[e] > ex[i0]) i0 = e;
    int i1 = (i0 == 0) ? 1 : 0;
    for (int e = 0; e < NEXP; ++e) if (e != i0 && ex[e] > ex[i1]) i1 = e;
    float p0 = ex[i0], p1 = ex[i1], tsum = p0 + p1;
    sel[2 * row] = i0; sel[2 * row + 1] = i1;
    selw[2 * row] = p0 / tsum; selw[2 * row + 1] = p1 / tsum;
    atomicAdd(&counts[i0], 1);
    atomicAdd(&counts[i1], 1);
  }
}

__global__ void zero_small_kernel(int* counts, int* cursor) {
  int t = threadIdx.x;
  if (t < NEXP) { counts[t] = 0; cursor[t] = 0; }
}

__global__ void offs_kernel(const int* counts, int* offs, int* cursor) {
  if (threadIdx.x == 0) {
    int a = 0;
    for (int e = 0; e < NEXP; ++e) { offs[e] = a; a += counts[e]; cursor[e] = 0; }
  }
}

__global__ void scatter_kernel(const int* sel, const float* selw, const int* offs, int* cursor,
                               int* gidx, float* gw, int* pos) {
  int t = blockIdx.x * blockDim.x + threadIdx.x;
  if (t < T_TOK) {
#pragma unroll
    for (int k = 0; k < 2; ++k) {
      int e = sel[2 * t + k];
      int p = atomicAdd(&cursor[e], 1);
      int g = offs[e] + p;
      gidx[g] = t; gw[g] = selw[2 * t + k]; pos[2 * t + k] = g;
    }
  }
}

__global__ void hcomb_kernel(const float* __restrict__ t1, const float* __restrict__ t3,
                             __bf16* __restrict__ Hbf) {
  size_t base = (size_t)blockIdx.x * 2048 + threadIdx.x * 4;
#pragma unroll
  for (int i = 0; i < 2; ++i) {
    size_t idx = base + i * 1024;
    float4 a = *(const float4*)(t1 + idx);
    float4 c = *(const float4*)(t3 + idx);
    st_bf16x4(Hbf + idx,
              (a.x / (1.0f + __expf(-a.x))) * c.x,
              (a.y / (1.0f + __expf(-a.y))) * c.y,
              (a.z / (1.0f + __expf(-a.z))) * c.z,
              (a.w / (1.0f + __expf(-a.w))) * c.w);
  }
}

__global__ void combine_kernel(const float* __restrict__ Yp, const int* __restrict__ pos,
                               float* __restrict__ out) {
  int row = blockIdx.x;
  int g0 = pos[2 * row], g1 = pos[2 * row + 1];
  const float* y0 = Yp + (size_t)g0 * HID;
  const float* y1 = Yp + (size_t)g1 * HID;
  float* o = out + (size_t)row * HID;
#pragma unroll
  for (int i = 0; i < 2; ++i) {
    int j = threadIdx.x * 4 + i * 1024;
    float4 ov = *(const float4*)(o + j);
    float4 a = *(const float4*)(y0 + j);
    float4 b = *(const float4*)(y1 + j);
    *(float4*)(o + j) = make_float4(ov.x + a.x + b.x, ov.y + a.y + b.y,
                                    ov.z + a.z + b.z, ov.w + a.w + b.w);
  }
}

// ---------------------------------------------------------------------------
// Launch
// ---------------------------------------------------------------------------
extern "C" void kernel_launch(void* const* d_in, const int* in_sizes, int n_in,
                              void* d_out, int out_size, void* d_ws, size_t ws_size,
                              hipStream_t stream) {
  (void)in_sizes; (void)n_in; (void)out_size; (void)ws_size;
  const float* x        = (const float*)d_in[0];
  const float* slope    = (const float*)d_in[1];
  const float* ln1w     = (const float*)d_in[2];
  const float* ln2w     = (const float*)d_in[3];
  const float* qkv_w    = (const float*)d_in[4];
  const float* ogate_w  = (const float*)d_in[5];
  const float* anorm_w  = (const float*)d_in[6];
  const float* aout_w   = (const float*)d_in[7];
  const float* router_w = (const float*)d_in[8];
  const float* w1       = (const float*)d_in[9];
  const float* w2       = (const float*)d_in[10];
  const float* w3       = (const float*)d_in[11];
  float* out = (float*)d_out;

  char* ws = (char*)d_ws;
  float*  h_ln1 = (float*)(ws + (0ull   << 20));  //  8 MiB [1024,2048]
  float*  qkv   = (float*)(ws + (8ull   << 20));  // 24 MiB [1024,6144]
  float*  gate  = (float*)(ws + (32ull  << 20));  //  8 MiB
  float*  kvb   = (float*)(ws + (40ull  << 20));  //  4 MiB [16,4,128,128]
  float*  kvs   = (float*)(ws + (44ull  << 20));  //  4 MiB
  float*  attn  = (float*)(ws + (48ull  << 20));  //  8 MiB
  float*  m_in  = (float*)(ws + (56ull  << 20));  //  8 MiB
  float*  x2    = (float*)(ws + (64ull  << 20));  //  8 MiB
  float*  t1    = (float*)(ws + (72ull  << 20));  // 16 MiB [2048,2048]
  float*  t3    = (float*)(ws + (88ull  << 20));  // 16 MiB
  __bf16* Hbf   = (__bf16*)(ws + (104ull << 20)); //  8 MiB
  float*  Yp    = (float*)(ws + (112ull << 20));  // 16 MiB
  char*   misc  = ws + (128ull << 20);
  int*    counts = (int*)(misc + 0);
  int*    offs   = (int*)(misc + 64);
  int*    cursor = (int*)(misc + 128);
  int*    sel    = (int*)(misc + 256);
  int*    pos    = (int*)(misc + 256 + 8192);
  int*    gidx   = (int*)(misc + 256 + 16384);
  float*  selw   = (float*)(misc + 256 + 24576);
  float*  gw     = (float*)(misc + 256 + 32768);

  zero_small_kernel<<<1, 32, 0, stream>>>(counts, cursor);

  // ln1 -> h ; qkv = silu(h@qkv_w^T) ; gate = sigmoid(h@ogate_w^T)
  rmsnorm_kernel<<<T_TOK, 256, 0, stream>>>(x, ln1w, EPS_LN, h_ln1);
  gemm_kernel<EPI_SILU, false, false, false><<<dim3(48, 8, 1), 256, 0, stream>>>(
      h_ln1, qkv_w, qkv, T_TOK, QKVN, HID, HID, QKVN, nullptr, nullptr, nullptr, nullptr, 0);
  gemm_kernel<EPI_SIGMOID, false, false, false><<<dim3(16, 8, 1), 256, 0, stream>>>(
      h_ln1, ogate_w, gate, T_TOK, HID, HID, HID, HID, nullptr, nullptr, nullptr, nullptr, 0);

  // lightning attention
  attn_kv_kernel<<<dim3(NBLK, NH, 1), 256, 0, stream>>>(qkv, slope, kvb);
  attn_prefix_kernel<<<dim3(NBLK, NH, 1), 256, 0, stream>>>(kvb, slope, kvs);
  attn_out_kernel<<<dim3(NBLK, NH, 1), 256, 0, stream>>>(qkv, kvs, slope, attn);

  // rmsnorm * gate -> out-proj + residual into d_out
  attn_gate_kernel<<<T_TOK, 256, 0, stream>>>(attn, anorm_w, gate, m_in);
  gemm_kernel<EPI_RESADD, false, false, false><<<dim3(16, 8, 1), 256, 0, stream>>>(
      m_in, aout_w, out, T_TOK, HID, HID, HID, HID, x, nullptr, nullptr, nullptr, 0);

  // ln2 + router + top-2 routing
  ln2_router_kernel<<<T_TOK, 256, 0, stream>>>(out, ln2w, router_w, x2, sel, selw, counts);
  offs_kernel<<<1, 32, 0, stream>>>(counts, offs, cursor);
  scatter_kernel<<<4, 256, 0, stream>>>(sel, selw, offs, cursor, gidx, gw, pos);

  // grouped expert GEMMs (sparse top-2 == reference dense-then-mask)
  gemm_kernel<EPI_NONE, false, true, true><<<dim3(16, 16, 8), 256, 0, stream>>>(
      x2, w1, t1, 0, FFN, HID, HID, FFN, nullptr, gidx, counts, offs, (long long)FFN * HID);
  gemm_kernel<EPI_NONE, false, true, true><<<dim3(16, 16, 8), 256, 0, stream>>>(
      x2, w3, t3, 0, FFN, HID, HID, FFN, nullptr, gidx, counts, offs, (long long)FFN * HID);
  hcomb_kernel<<<2048, 256, 0, stream>>>(t1, t3, Hbf);
  gemm_kernel<EPI_ROWSCALE, true, true, false><<<dim3(16, 16, 8), 256, 0, stream>>>(
      Hbf, w2, Yp, 0, HID, FFN, FFN, HID, gw, nullptr, counts, offs, (long long)HID * FFN);

  // out += the token's two expert partials (deterministic, no float atomics)
  combine_kernel<<<T_TOK, 256, 0, stream>>>(Yp, pos, out);
}
